// MultiHeadAttention_63118839382316
// MI455X (gfx1250) — compile-verified, matrix operand order checked by reading, not run on hardware
//
#include <hip/hip_runtime.h>
#include <math.h>
#include <stdint.h>

// ---------------------------------------------------------------------------
// Causal multi-head attention forward for MI455X (gfx1250, wave32, WMMA).
//   B=2, T=4096, H=8, D=64, C=512.
// All matmuls via v_wmma_f32_16x16x32_bf16 (f32 accumulate).
// K-tile staging in attention uses global_load_async_to_lds_b128 (ASYNCcnt),
// next-tile HBM traffic is warmed with global_prefetch_b8.
// Workspace: Q,K,V bf16 [B][H][T][D] + O bf16 [B][T][C]  = 32 MB total.
// ---------------------------------------------------------------------------

#define HEADS    8
#define HEAD_DIM 64
#define DIM      512
#define NQKV     (3 * DIM)
#define TSEQ     4096
#define BATCH    2
#define MROWS    (BATCH * TSEQ)   // 8192

typedef __attribute__((ext_vector_type(16))) __bf16 bf16x16;
typedef __attribute__((ext_vector_type(8)))  __bf16 bf16x8;
typedef __attribute__((ext_vector_type(8)))  float  f32x8;

// D = A(16x32 bf16) * B(32x16 bf16) + C(16x16 f32)
__device__ __forceinline__ f32x8 wmma_bf16(bf16x16 a, bf16x16 b, f32x8 c) {
  return __builtin_amdgcn_wmma_f32_16x16x32_bf16(false, a, false, b,
                                                 (short)0, c, false, false);
}

// Load one 16x32 bf16 WMMA fragment from an LDS tile stored row-major
// ([row][k], row stride `stride` elems). Works for A fragments (row=M) and
// B fragments when the tile is stored [N][K] ("column-major B").
// Lane L -> row = L%16; half-wave (L/16) selects K-offset 8;
// VGPR0..3 = K kh..kh+7, VGPR4..7 = K kh+16..kh+23.  => two 16B ds loads.
__device__ __forceinline__ bf16x16 load_frag(const __bf16* base, int stride,
                                             int row0, int koff, int lane) {
  const __bf16* p = base + (row0 + (lane & 15)) * stride + koff + ((lane >> 4) << 3);
  bf16x8 lo = *(const bf16x8*)p;
  bf16x8 hi = *(const bf16x8*)(p + 16);
  bf16x16 r;
#pragma unroll
  for (int u = 0; u < 8; ++u) { r[u] = lo[u]; r[8 + u] = hi[u]; }
  return r;
}

__device__ __forceinline__ f32x8 zero8() {
  f32x8 z;
#pragma unroll
  for (int u = 0; u < 8; ++u) z[u] = 0.0f;
  return z;
}

// Async 16B global->LDS copy (CDNA5 GVS mode, tracked with ASYNCcnt).
// lds_off = wave-relative LDS byte address (low 32 bits of generic pointer,
// per the aperture mapping).
__device__ __forceinline__ void async_copy_b128(const __bf16* ldsp,
                                                const __bf16* gbase,
                                                unsigned byte_off) {
  unsigned lds_off = (unsigned)(uintptr_t)(const void*)ldsp;
  asm volatile("global_load_async_to_lds_b128 %0, %1, %2"
               :: "v"(lds_off), "v"(byte_off),
                  "s"((unsigned long long)(uintptr_t)gbase)
               : "memory");
}
__device__ __forceinline__ void async_wait0() {
  asm volatile("s_wait_asynccnt 0" ::: "memory");
}

// ---------------------------------------------------------------------------
// Kernel 1: qkv = x @ w_qkv + b_qkv, scattered to Q/K/V bf16 [B][H][T][D].
// Block tile 128x64 (4 waves, each owns a 32x64 strip -> 8 WMMAs / K-step).
// ---------------------------------------------------------------------------
__global__ __launch_bounds__(128)
void qkv_gemm_kernel(const float* __restrict__ x, const float* __restrict__ w,
                     const float* __restrict__ bias,
                     __bf16* __restrict__ Q, __bf16* __restrict__ K,
                     __bf16* __restrict__ V) {
  __shared__ __bf16 As[128 * 40];  // x tile  [m][k], k-chunk of 32 (+8 pad)
  __shared__ __bf16 Bs[64 * 40];   // w tile  [n][k] (transposed on stage)
  const int t = threadIdx.x, lane = t & 31, wv = t >> 5;
  const int mbase = blockIdx.y * 128;
  const int ntile = blockIdx.x * 64;

  f32x8 acc[2][4];
#pragma unroll
  for (int mi = 0; mi < 2; ++mi)
#pragma unroll
    for (int i = 0; i < 4; ++i) acc[mi][i] = zero8();

  for (int kk = 0; kk < DIM; kk += 32) {
    // stage A: 128x32 f32 -> bf16, coalesced float4 reads
#pragma unroll
    for (int e = 0; e < 8; ++e) {
      int l4 = e * 128 + t;
      int row = l4 >> 3, c = (l4 & 7) * 4;
      float4 f = *(const float4*)(x + (size_t)(mbase + row) * DIM + kk + c);
      __bf16* d = As + row * 40 + c;
      d[0] = (__bf16)f.x; d[1] = (__bf16)f.y; d[2] = (__bf16)f.z; d[3] = (__bf16)f.w;
    }
    // stage B transposed: w[k][n] -> Bs[n][k]
#pragma unroll
    for (int e = 0; e < 4; ++e) {
      int l4 = e * 128 + t;
      int k = l4 >> 4, n = (l4 & 15) * 4;
      float4 f = *(const float4*)(w + (size_t)(kk + k) * NQKV + ntile + n);
      Bs[(n + 0) * 40 + k] = (__bf16)f.x;
      Bs[(n + 1) * 40 + k] = (__bf16)f.y;
      Bs[(n + 2) * 40 + k] = (__bf16)f.z;
      Bs[(n + 3) * 40 + k] = (__bf16)f.w;
    }
    __syncthreads();
    bf16x16 a0 = load_frag(As, 40, wv * 32, 0, lane);
    bf16x16 a1 = load_frag(As, 40, wv * 32 + 16, 0, lane);
#pragma unroll
    for (int ns = 0; ns < 4; ++ns) {
      bf16x16 bfr = load_frag(Bs, 40, ns * 16, 0, lane);
      acc[0][ns] = wmma_bf16(a0, bfr, acc[0][ns]);
      acc[1][ns] = wmma_bf16(a1, bfr, acc[1][ns]);
    }
    __syncthreads();
  }

  // C/D layout: VGPR r, lane L -> M = r + 8*(L/16), N = L%16
#pragma unroll
  for (int mi = 0; mi < 2; ++mi)
#pragma unroll
    for (int ns = 0; ns < 4; ++ns)
#pragma unroll
      for (int r = 0; r < 8; ++r) {
        int m = mbase + wv * 32 + mi * 16 + r + ((lane >> 4) << 3);
        int n = ntile + ns * 16 + (lane & 15);
        float v = acc[mi][ns][r] + bias[n];
        int part = n >> 9, c = n & 511, h = c >> 6, d = c & 63;
        int b = m >> 12, tt = m & (TSEQ - 1);
        __bf16* dst = (part == 0) ? Q : (part == 1) ? K : V;
        dst[((size_t)(b * HEADS + h) * TSEQ + tt) * HEAD_DIM + d] = (__bf16)v;
      }
}

// ---------------------------------------------------------------------------
// Kernel 2: causal flash attention over one (b,h) head, 64-query tile/block.
// Grid: (T/64, B*H), block 128 (4 waves x 16 query rows).
// ---------------------------------------------------------------------------
__global__ __launch_bounds__(128)
void attn_kernel(const __bf16* __restrict__ Q, const __bf16* __restrict__ K,
                 const __bf16* __restrict__ V, __bf16* __restrict__ O) {
  __shared__ __bf16 Qs[64 * 72];        // [q][d], pre-scaled by 1/sqrt(D)
  __shared__ __bf16 Ks[64 * 72];        // [key][d]  == B layout for Q*K^T
  __shared__ __bf16 Vt[64 * 72];        // [d][key]  == B layout for P*V
  __shared__ __bf16 Ps[4 * 16 * 72];    // per-wave P tile [q][key] bf16

  const int t = threadIdx.x, lane = t & 31, wv = t >> 5;
  const int bh = blockIdx.y, b = bh >> 3, h = bh & 7;
  const int iq = blockIdx.x, qbase = iq * 64;
  const __bf16* Qh = Q + (size_t)bh * TSEQ * HEAD_DIM;
  const __bf16* Kh = K + (size_t)bh * TSEQ * HEAD_DIM;
  const __bf16* Vh = V + (size_t)bh * TSEQ * HEAD_DIM;

  // stage Q tile once, folding in softmax scale 1/8 (exact in bf16)
#pragma unroll
  for (int e = 0; e < 4; ++e) {
    int l8 = e * 128 + t;
    int row = l8 >> 3, c = (l8 & 7) * 8;
    bf16x8 v = *(const bf16x8*)(Qh + (size_t)(qbase + row) * HEAD_DIM + c);
    bf16x8 o;
#pragma unroll
    for (int u = 0; u < 8; ++u) o[u] = (__bf16)((float)v[u] * 0.125f);
    *(bf16x8*)(Qs + row * 72 + c) = o;
  }

  float mrun[8], lrun[8];
  f32x8 accO[4];
#pragma unroll
  for (int r = 0; r < 8; ++r) { mrun[r] = -1e30f; lrun[r] = 0.0f; }
#pragma unroll
  for (int ns = 0; ns < 4; ++ns) accO[ns] = zero8();

  for (int j = 0; j <= iq; ++j) {
    __syncthreads();   // prior PV reads done (also covers Q staging, iter 0)

    // stage K tile [key][d] via async global->LDS (CDNA5 path, ASYNCcnt)
#pragma unroll
    for (int e = 0; e < 4; ++e) {
      int l8 = e * 128 + t;
      int row = l8 >> 3, c = (l8 & 7) * 8;
      async_copy_b128(Ks + row * 72 + c, Kh,
                      (unsigned)(((j * 64 + row) * HEAD_DIM + c) * 2));
    }
    // stage V transposed [d][key] (through VGPRs: transpose on LDS store)
#pragma unroll
    for (int e = 0; e < 4; ++e) {
      int l8 = e * 128 + t;
      int key = l8 >> 3, d0 = (l8 & 7) * 8;
      bf16x8 v = *(const bf16x8*)(Vh + (size_t)(j * 64 + key) * HEAD_DIM + d0);
#pragma unroll
      for (int u = 0; u < 8; ++u) Vt[(d0 + u) * 72 + key] = v[u];
    }
    // warm HBM/L2 for the next key tile while we compute on this one
    if (j < iq) {
      const __bf16* nK = Kh + (size_t)(j + 1) * 64 * HEAD_DIM;
      const __bf16* nV = Vh + (size_t)(j + 1) * 64 * HEAD_DIM;
      __builtin_prefetch(nK + t * 32, 0, 1);   // 128 thr * 64B = full 8KB tile
      __builtin_prefetch(nV + t * 32, 0, 1);
    }
    async_wait0();
    __syncthreads();

    // S = Q * K^T   (16 q-rows x 64 keys per wave, 8 WMMAs)
    f32x8 s[4];
#pragma unroll
    for (int ns = 0; ns < 4; ++ns) s[ns] = zero8();
#pragma unroll
    for (int kc = 0; kc < HEAD_DIM; kc += 32) {
      bf16x16 aq = load_frag(Qs, 72, wv * 16, kc, lane);
#pragma unroll
      for (int ns = 0; ns < 4; ++ns) {
        bf16x16 bk = load_frag(Ks, 72, ns * 16, kc, lane);
        s[ns] = wmma_bf16(aq, bk, s[ns]);
      }
    }

    // causal mask (only the diagonal key tile needs it)
    if (j == iq) {
#pragma unroll
      for (int ns = 0; ns < 4; ++ns)
#pragma unroll
        for (int r = 0; r < 8; ++r) {
          int qg = qbase + wv * 16 + r + ((lane >> 4) << 3);
          int kg = j * 64 + ns * 16 + (lane & 15);
          s[ns][r] = (kg > qg) ? -1e30f : s[ns][r];
        }
    }

    // online softmax; a row's 16 cols live across one 16-lane group
#pragma unroll
    for (int r = 0; r < 8; ++r) {
      float mx = fmaxf(fmaxf(s[0][r], s[1][r]), fmaxf(s[2][r], s[3][r]));
#pragma unroll
      for (int off = 8; off; off >>= 1) mx = fmaxf(mx, __shfl_xor(mx, off, 32));
      float mnew  = fmaxf(mrun[r], mx);
      float alpha = __expf(mrun[r] - mnew);
      float rs = 0.0f;
      int prow = wv * 16 * 72 + (r + ((lane >> 4) << 3)) * 72 + (lane & 15);
#pragma unroll
      for (int ns = 0; ns < 4; ++ns) {
        float p = __expf(s[ns][r] - mnew);
        rs += p;
        Ps[prow + ns * 16] = (__bf16)p;      // P in A-fragment-friendly [q][key]
      }
#pragma unroll
      for (int off = 8; off; off >>= 1) rs += __shfl_xor(rs, off, 32);
      lrun[r] = lrun[r] * alpha + rs;
      mrun[r] = mnew;
#pragma unroll
      for (int ns = 0; ns < 4; ++ns) accO[ns][r] *= alpha;
    }
    __syncthreads();

    // O += P * V   (8 WMMAs)
#pragma unroll
    for (int kc = 0; kc < 64; kc += 32) {
      bf16x16 ap = load_frag(Ps + wv * 16 * 72, 72, 0, kc, lane);
#pragma unroll
      for (int ns = 0; ns < 4; ++ns) {
        bf16x16 bv = load_frag(Vt, 72, ns * 16, kc, lane);
        accO[ns] = wmma_bf16(ap, bv, accO[ns]);
      }
    }
  }

  // normalize and store O as bf16 [B][T][C] (row-major for the proj GEMM)
#pragma unroll
  for (int r = 0; r < 8; ++r) {
    float inv = 1.0f / lrun[r];
    int tt = qbase + wv * 16 + r + ((lane >> 4) << 3);
#pragma unroll
    for (int ns = 0; ns < 4; ++ns) {
      int c = h * HEAD_DIM + ns * 16 + (lane & 15);
      O[((size_t)(b * TSEQ + tt)) * DIM + c] = (__bf16)(accO[ns][r] * inv);
    }
  }
}

// ---------------------------------------------------------------------------
// Kernel 3: out = O @ w_proj + b_proj  (f32 output), block tile 128x64.
// ---------------------------------------------------------------------------
__global__ __launch_bounds__(128)
void proj_gemm_kernel(const __bf16* __restrict__ A, const float* __restrict__ w,
                      const float* __restrict__ bias, float* __restrict__ out) {
  __shared__ __bf16 As[128 * 40];
  __shared__ __bf16 Bs[64 * 40];
  const int t = threadIdx.x, lane = t & 31, wv = t >> 5;
  const int mbase = blockIdx.y * 128;
  const int ntile = blockIdx.x * 64;

  f32x8 acc[2][4];
#pragma unroll
  for (int mi = 0; mi < 2; ++mi)
#pragma unroll
    for (int i = 0; i < 4; ++i) acc[mi][i] = zero8();

  for (int kk = 0; kk < DIM; kk += 32) {
#pragma unroll
    for (int e = 0; e < 4; ++e) {          // A already bf16: 16B copies
      int l8 = e * 128 + t;
      int row = l8 >> 2, c = (l8 & 3) * 8;
      *(bf16x8*)(As + row * 40 + c) =
          *(const bf16x8*)(A + (size_t)(mbase + row) * DIM + kk + c);
    }
#pragma unroll
    for (int e = 0; e < 4; ++e) {          // w_proj f32 -> bf16, transposed
      int l4 = e * 128 + t;
      int k = l4 >> 4, n = (l4 & 15) * 4;
      float4 f = *(const float4*)(w + (size_t)(kk + k) * DIM + ntile + n);
      Bs[(n + 0) * 40 + k] = (__bf16)f.x;
      Bs[(n + 1) * 40 + k] = (__bf16)f.y;
      Bs[(n + 2) * 40 + k] = (__bf16)f.z;
      Bs[(n + 3) * 40 + k] = (__bf16)f.w;
    }
    __syncthreads();
    bf16x16 a0 = load_frag(As, 40, wv * 32, 0, lane);
    bf16x16 a1 = load_frag(As, 40, wv * 32 + 16, 0, lane);
#pragma unroll
    for (int ns = 0; ns < 4; ++ns) {
      bf16x16 bfr = load_frag(Bs, 40, ns * 16, 0, lane);
      acc[0][ns] = wmma_bf16(a0, bfr, acc[0][ns]);
      acc[1][ns] = wmma_bf16(a1, bfr, acc[1][ns]);
    }
    __syncthreads();
  }

#pragma unroll
  for (int mi = 0; mi < 2; ++mi)
#pragma unroll
    for (int ns = 0; ns < 4; ++ns)
#pragma unroll
      for (int r = 0; r < 8; ++r) {
        int m = mbase + wv * 32 + mi * 16 + r + ((lane >> 4) << 3);
        int n = ntile + ns * 16 + (lane & 15);
        out[(size_t)m * DIM + n] = acc[mi][ns][r] + bias[n];
      }
}

// ---------------------------------------------------------------------------
extern "C" void kernel_launch(void* const* d_in, const int* in_sizes, int n_in,
                              void* d_out, int out_size, void* d_ws, size_t ws_size,
                              hipStream_t stream) {
  (void)in_sizes; (void)n_in; (void)out_size; (void)ws_size;
  const float* x      = (const float*)d_in[0];
  const float* w_qkv  = (const float*)d_in[1];
  const float* b_qkv  = (const float*)d_in[2];
  const float* w_proj = (const float*)d_in[3];
  const float* b_proj = (const float*)d_in[4];
  float* out = (float*)d_out;

  // workspace: Q,K,V bf16 [B][H][T][D] (8MB each) + O bf16 [B][T][C] (8MB)
  const size_t per = (size_t)BATCH * HEADS * TSEQ * HEAD_DIM;  // 4M elems
  char* ws = (char*)d_ws;
  __bf16* Q = (__bf16*)(ws);
  __bf16* K = (__bf16*)(ws + 2 * per);
  __bf16* V = (__bf16*)(ws + 4 * per);
  __bf16* O = (__bf16*)(ws + 6 * per);

  qkv_gemm_kernel<<<dim3(NQKV / 64, MROWS / 128), 128, 0, stream>>>(
      x, w_qkv, b_qkv, Q, K, V);
  attn_kernel<<<dim3(TSEQ / 64, BATCH * HEADS), 128, 0, stream>>>(Q, K, V, O);
  proj_gemm_kernel<<<dim3(DIM / 64, MROWS / 128), 128, 0, stream>>>(
      O, w_proj, b_proj, out);
}
